// GraphConvolutionNetwork_53291954208986
// MI455X (gfx1250) — matrix operand order repaired and checked by reading, not compiled, not run on hardware
//
#include <hip/hip_runtime.h>
#include <hip/hip_bf16.h>

#define N0_NODES 100000
#define K1_NODES 50000
#define K2_NODES 25000
#define N_EDGES  3200000

typedef __attribute__((ext_vector_type(16))) __bf16          v16bf;
typedef __attribute__((ext_vector_type(8)))  unsigned short  v8us;
typedef __attribute__((ext_vector_type(8)))  float           v8f;

union BFrag {                       // 16 bf16 elements = 8 VGPRs
    struct { v8us lo, hi; } p;      // two 16-byte halves (for LDS b128 loads)
    v16bf b;
};

struct PoolState { float lo, hi, mid; int count; };

// ---------------------------------------------------------------------------
// WMMA GEMM: H[n x (NT*16)] = X[n x 64] @ W[64 x (NT*16)]   (bf16 in, f32 acc)
//  - one wave per 16-row tile, K=64 as two K32 chunks -> v_wmma_f32_16x16x32_bf16
//  - A: four global_load_b128 per chunk (row clamped, no per-element guards)
//  - B: W transposed->bf16 staged in LDS once per block; two ds_load_b128/frag
//  - C: wave-uniform fast-path store for full tiles (no exec-mask toggles)
// ---------------------------------------------------------------------------
template <int NT>
__global__ __launch_bounds__(256) void gemm_wmma(const float* __restrict__ X,
                                                 const float* __restrict__ W,
                                                 float* __restrict__ H, int n)
{
    constexpr int DOUT = NT * 16;
    __shared__ __bf16 WT[DOUT * 64];          // WT[c*64 + k] = bf16(W[k][c])

    // ---- cooperative stage of W (transposed, bf16) into LDS
    for (int idx = threadIdx.x; idx < DOUT * 64; idx += 256) {
        const int c = idx >> 6;               // output column
        const int k = idx & 63;               // K index
        WT[idx] = (__bf16)W[(size_t)k * DOUT + c];
    }
    __syncthreads();

    const int lane = threadIdx.x & 31;
    const int wave = threadIdx.x >> 5;
    const int tile = blockIdx.x * 8 + wave;
    const int row0 = tile * 16;
    if (row0 >= n) return;                    // wave-uniform exit

    const int m    = lane & 15;               // A row / C col within tile
    const int hg   = lane >> 4;               // half-group: 0 or 1
    const int row  = row0 + m;
    const int srow = (row < n) ? row : (n - 1);   // clamp: OOB rows never stored
    const float* __restrict__ xr = X + (size_t)srow * 64;
    __builtin_prefetch(xr, 0, 0);             // global_prefetch_b8

    const int kbase = hg * 8;                 // A K-offset per half-group
    const int kb2   = hg * 8;                 // B K-offset per half-group

    v8f acc[NT];
#pragma unroll
    for (int t = 0; t < NT; ++t)
        acc[t] = (v8f){0.f, 0.f, 0.f, 0.f, 0.f, 0.f, 0.f, 0.f};

#pragma unroll
    for (int kk = 0; kk < 64; kk += 32) {
        // ---- A fragment: elements 0..7 <-> K=kbase+e, 8..15 <-> K=16+kbase+e-8
        const float4 f0 = *(const float4*)(xr + kk + kbase);
        const float4 f1 = *(const float4*)(xr + kk + kbase + 4);
        const float4 f2 = *(const float4*)(xr + kk + kbase + 16);
        const float4 f3 = *(const float4*)(xr + kk + kbase + 20);
        v16bf A;
        A[0]  = (__bf16)f0.x; A[1]  = (__bf16)f0.y; A[2]  = (__bf16)f0.z; A[3]  = (__bf16)f0.w;
        A[4]  = (__bf16)f1.x; A[5]  = (__bf16)f1.y; A[6]  = (__bf16)f1.z; A[7]  = (__bf16)f1.w;
        A[8]  = (__bf16)f2.x; A[9]  = (__bf16)f2.y; A[10] = (__bf16)f2.z; A[11] = (__bf16)f2.w;
        A[12] = (__bf16)f3.x; A[13] = (__bf16)f3.y; A[14] = (__bf16)f3.z; A[15] = (__bf16)f3.w;

        // ---- B fragments: lane holds col, K=kb2+0..7 and 16+kb2+0..7 -> two ds_load_b128
#pragma unroll
        for (int t = 0; t < NT; ++t) {
            const int col = t * 16 + m;
            const __bf16* wp = WT + col * 64 + kk + kb2;   // 16B-aligned
            BFrag B;
            B.p.lo = *(const v8us*)wp;
            B.p.hi = *(const v8us*)(wp + 16);
            acc[t] = __builtin_amdgcn_wmma_f32_16x16x32_bf16(
                false, A, false, B.b, (short)0, acc[t], false, false);
        }
    }

    // ---- store C/D: VGPR r -> M = r + 8*hg, N = m
    float* __restrict__ hb = H + (size_t)(row0 + hg * 8) * DOUT + m;
    if (row0 + 16 <= n) {
        // full tile (the common case): no guards, no exec-mask churn
#pragma unroll
        for (int t = 0; t < NT; ++t) {
#pragma unroll
            for (int r = 0; r < 8; ++r)
                hb[(size_t)r * DOUT + t * 16] = acc[t][r];
        }
    } else {
        // tail tile only (K2 = 25000 is not a multiple of 16)
#pragma unroll
        for (int t = 0; t < NT; ++t) {
#pragma unroll
            for (int r = 0; r < 8; ++r) {
                const int rr = row0 + hg * 8 + r;
                if (rr < n) H[(size_t)rr * DOUT + t * 16 + m] = acc[t][r];
            }
        }
    }
}

// ---------------------------------------------------------------------------
// Utility fills
// ---------------------------------------------------------------------------
__global__ void fill_f32(float* p, float v, int n) {
    int i = blockIdx.x * blockDim.x + threadIdx.x;
    if (i < n) p[i] = v;
}
__global__ void fill_i32(int* p, int v, int n) {
    int i = blockIdx.x * blockDim.x + threadIdx.x;
    if (i < n) p[i] = v;
}
__global__ void init_edges(const int* __restrict__ ei, int* s, int* d, float* w, int E) {
    int i = blockIdx.x * blockDim.x + threadIdx.x;
    if (i >= E) return;
    s[i] = ei[i];
    d[i] = ei[E + i];
    w[i] = 1.0f;
}

// ---------------------------------------------------------------------------
// GCN pieces
// ---------------------------------------------------------------------------
__global__ void degree_kernel(const int* __restrict__ dst, const float* __restrict__ ew,
                              float* __restrict__ deg, int E) {
    int i = blockIdx.x * blockDim.x + threadIdx.x;
    if (i >= E) return;
    float w = ew[i];
    if (w != 0.0f) atomicAdd(&deg[dst[i]], w);
}
__global__ void to_dis(float* deg, int n) {
    int i = blockIdx.x * blockDim.x + threadIdx.x;
    if (i < n) deg[i] = rsqrtf(deg[i] + 1.0f);   // +1 self loop; always > 0
}
template <int DOUT>
__global__ void aggregate_kernel(const float* __restrict__ h, const int* __restrict__ src,
                                 const int* __restrict__ dst, const float* __restrict__ ew,
                                 const float* __restrict__ dis, float* __restrict__ agg,
                                 int E) {
    constexpr int FP = DOUT / 16;                // features per thread
    long long t = (long long)blockIdx.x * blockDim.x + threadIdx.x;
    int e = (int)(t >> 4);
    if (e >= E) return;
    int q = (int)(t & 15);
    float w = ew[e];
    if (w == 0.0f) return;
    int s = src[e], d = dst[e];
    float norm = dis[s] * w * dis[d];
    const float* __restrict__ hs = h + (size_t)s * DOUT + q * FP;
    float* __restrict__ ad       = agg + (size_t)d * DOUT + q * FP;
#pragma unroll
    for (int j = 0; j < FP; ++j)
        atomicAdd(&ad[j], hs[j] * norm);
}
template <int DOUT>
__global__ void finalize_kernel(const float* __restrict__ h, const float* __restrict__ agg,
                                const float* __restrict__ dis, const float* __restrict__ bias,
                                float* __restrict__ out, int n, int relu) {
    int t = blockIdx.x * blockDim.x + threadIdx.x;
    if (t >= n * DOUT) return;
    int i = t / DOUT;                // power-of-two -> shift
    int d = t & (DOUT - 1);
    float di = dis[i];
    float v  = agg[t] + h[t] * di * di + bias[d];
    if (relu) v = fmaxf(v, 0.0f);
    out[t] = v;
}

// ---------------------------------------------------------------------------
// TopK pooling pieces (device-side bisection + deterministic compaction)
// ---------------------------------------------------------------------------
__global__ void score_kernel(const float* __restrict__ x, const float* __restrict__ p,
                             float* __restrict__ score, int n) {
    int lane = threadIdx.x & 31;
    int wave = threadIdx.x >> 5;
    int node = blockIdx.x * 8 + wave;
    if (node >= n) return;
    float p0 = p[lane], p1 = p[lane + 32];
    float v0 = x[(size_t)node * 64 + lane];
    float v1 = x[(size_t)node * 64 + lane + 32];
    float dot = v0 * p0 + v1 * p1;
    float nn  = p0 * p0 + p1 * p1;
    for (int m = 16; m >= 1; m >>= 1) {
        dot += __shfl_xor(dot, m, 32);
        nn  += __shfl_xor(nn,  m, 32);
    }
    if (lane == 0) score[node] = tanhf(dot * rsqrtf(nn));
}
__global__ void pool_init(PoolState* st) {
    st->lo = -1.0001f; st->hi = 1.0001f; st->mid = 0.0f; st->count = 0;
}
__global__ void pool_count(const float* __restrict__ score, int n, PoolState* st) {
    __shared__ int c;
    if (threadIdx.x == 0) c = 0;
    __syncthreads();
    int i = blockIdx.x * blockDim.x + threadIdx.x;
    float mid = st->mid;
    if (i < n && score[i] > mid) atomicAdd(&c, 1);
    __syncthreads();
    if (threadIdx.x == 0 && c) atomicAdd(&st->count, c);
}
__global__ void pool_update(PoolState* st, int K) {
    if (st->count >= K) st->lo = st->mid; else st->hi = st->mid;
    st->mid = 0.5f * (st->lo + st->hi);
    st->count = 0;
}
__global__ void block_count(const float* __restrict__ score, int n,
                            const PoolState* st, int* bc) {
    __shared__ int c;
    if (threadIdx.x == 0) c = 0;
    __syncthreads();
    int i = blockIdx.x * blockDim.x + threadIdx.x;
    float thr = st->lo;
    if (i < n && score[i] > thr) atomicAdd(&c, 1);
    __syncthreads();
    if (threadIdx.x == 0) bc[blockIdx.x] = c;
}
__global__ void scan_blocks(const int* __restrict__ bc, int* __restrict__ bo, int nb) {
    __shared__ int s[512];
    int t = threadIdx.x;
    int v = (t < nb) ? bc[t] : 0;
    s[t] = v;
    __syncthreads();
    for (int off = 1; off < 512; off <<= 1) {
        int u = (t >= off) ? s[t - off] : 0;
        __syncthreads();
        s[t] += u;
        __syncthreads();
    }
    if (t < nb) bo[t] = s[t] - v;   // exclusive prefix
}
__global__ void scatter_kernel(const float* __restrict__ x, const float* __restrict__ score,
                               const PoolState* st, int n, int K, const int* __restrict__ bo,
                               int* __restrict__ mapping, float* __restrict__ xk) {
    __shared__ int sc[256];
    int i = blockIdx.x * 256 + threadIdx.x;
    float thr = st->lo;
    int pred = (i < n && score[i] > thr) ? 1 : 0;
    sc[threadIdx.x] = pred;
    __syncthreads();
    for (int off = 1; off < 256; off <<= 1) {
        int u = (threadIdx.x >= off) ? sc[threadIdx.x - off] : 0;
        __syncthreads();
        sc[threadIdx.x] += u;
        __syncthreads();
    }
    int pos = bo[blockIdx.x] + sc[threadIdx.x] - pred;
    if (pred && pos < K) {
        mapping[i] = pos;
        float s = score[i];
        const float4* xs = (const float4*)(x + (size_t)i * 64);
        float4* xd = (float4*)(xk + (size_t)pos * 64);
#pragma unroll
        for (int d = 0; d < 16; ++d) {
            float4 v = xs[d];
            v.x *= s; v.y *= s; v.z *= s; v.w *= s;
            xd[d] = v;
        }
    }
}
__global__ void relabel_kernel(const int* __restrict__ si, const int* __restrict__ di,
                               const float* __restrict__ wi, const int* __restrict__ mapping,
                               int* __restrict__ so, int* __restrict__ dd,
                               float* __restrict__ wo, int E) {
    int i = blockIdx.x * blockDim.x + threadIdx.x;
    if (i >= E) return;
    int ns = mapping[si[i]];
    int nd = mapping[di[i]];
    float w = wi[i];
    bool valid = (ns >= 0) && (nd >= 0) && (w > 0.0f);
    so[i] = valid ? ns : 0;
    dd[i] = valid ? nd : 0;
    wo[i] = valid ? w : 0.0f;
}

// ---------------------------------------------------------------------------
// Host-side orchestration
// ---------------------------------------------------------------------------
static inline int cdiv(int a, int b) { return (a + b - 1) / b; }

static void run_gcn(const float* xin, const float* W, const float* bias,
                    const int* src, const int* dst, const float* ew,
                    int n, int dout, int relu,
                    float* h, float* agg, float* deg, float* out, hipStream_t stream)
{
    int tiles  = cdiv(n, 16);
    int blocks = cdiv(tiles, 8);
    fill_f32<<<cdiv(n, 256), 256, 0, stream>>>(deg, 0.0f, n);
    fill_f32<<<cdiv(n * dout, 256), 256, 0, stream>>>(agg, 0.0f, n * dout);
    if (dout == 64) gemm_wmma<4><<<blocks, 256, 0, stream>>>(xin, W, h, n);
    else            gemm_wmma<2><<<blocks, 256, 0, stream>>>(xin, W, h, n);

    degree_kernel<<<cdiv(N_EDGES, 256), 256, 0, stream>>>(dst, ew, deg, N_EDGES);
    to_dis<<<cdiv(n, 256), 256, 0, stream>>>(deg, n);

    long long tot = (long long)N_EDGES * 16;
    int ablocks = (int)((tot + 255) / 256);
    if (dout == 64)
        aggregate_kernel<64><<<ablocks, 256, 0, stream>>>(h, src, dst, ew, deg, agg, N_EDGES);
    else
        aggregate_kernel<32><<<ablocks, 256, 0, stream>>>(h, src, dst, ew, deg, agg, N_EDGES);

    if (dout == 64)
        finalize_kernel<64><<<cdiv(n * 64, 256), 256, 0, stream>>>(h, agg, deg, bias, out, n, relu);
    else
        finalize_kernel<32><<<cdiv(n * 32, 256), 256, 0, stream>>>(h, agg, deg, bias, out, n, relu);
}

static void run_pool(const float* xin, const float* p, int n, int K,
                     const int* si, const int* di, const float* wi,
                     int* so, int* dd, float* wo,
                     float* score, int* mapping, int* bc, int* bo, PoolState* st,
                     float* xk, hipStream_t stream)
{
    score_kernel<<<cdiv(n, 8), 256, 0, stream>>>(xin, p, score, n);
    pool_init<<<1, 1, 0, stream>>>(st);
    for (int it = 0; it < 40; ++it) {
        pool_count<<<cdiv(n, 256), 256, 0, stream>>>(score, n, st);
        pool_update<<<1, 1, 0, stream>>>(st, K);
    }
    fill_i32<<<cdiv(n, 256), 256, 0, stream>>>(mapping, -1, n);
    int nb = cdiv(n, 256);
    block_count<<<nb, 256, 0, stream>>>(score, n, st, bc);
    scan_blocks<<<1, 512, 0, stream>>>(bc, bo, nb);
    scatter_kernel<<<nb, 256, 0, stream>>>(xin, score, st, n, K, bo, mapping, xk);
    relabel_kernel<<<cdiv(N_EDGES, 256), 256, 0, stream>>>(si, di, wi, mapping, so, dd, wo, N_EDGES);
}

extern "C" void kernel_launch(void* const* d_in, const int* in_sizes, int n_in,
                              void* d_out, int out_size, void* d_ws, size_t ws_size,
                              hipStream_t stream)
{
    const float* x  = (const float*)d_in[0];
    const int*   ei = (const int*)d_in[1];
    // d_in[2] = batch (unused: single graph)
    const float* W0 = (const float*)d_in[3];  const float* b0 = (const float*)d_in[4];
    const float* W1 = (const float*)d_in[5];  const float* b1 = (const float*)d_in[6];
    const float* W2 = (const float*)d_in[7];  const float* b2 = (const float*)d_in[8];
    const float* W3 = (const float*)d_in[9];  const float* b3 = (const float*)d_in[10];
    const float* p0 = (const float*)d_in[11]; const float* p1 = (const float*)d_in[12];

    // ---- carve workspace
    char* ws = (char*)d_ws;
    auto carve = [&](size_t bytes) -> void* {
        void* p = (void*)ws;
        ws += (bytes + 255) & ~(size_t)255;
        return p;
    };
    float* nb0     = (float*)carve((size_t)N0_NODES * 64 * 4);
    float* nb1     = (float*)carve((size_t)N0_NODES * 64 * 4);
    float* agg     = (float*)carve((size_t)N0_NODES * 64 * 4);
    float* deg     = (float*)carve((size_t)N0_NODES * 4);
    float* score   = (float*)carve((size_t)N0_NODES * 4);
    int*   mapping = (int*)carve((size_t)N0_NODES * 4);
    int*   bc      = (int*)carve(512 * 4);
    int*   bo      = (int*)carve(512 * 4);
    PoolState* st  = (PoolState*)carve(sizeof(PoolState));
    int*   srcA    = (int*)carve((size_t)N_EDGES * 4);
    int*   dstA    = (int*)carve((size_t)N_EDGES * 4);
    float* ewA     = (float*)carve((size_t)N_EDGES * 4);
    int*   srcB    = (int*)carve((size_t)N_EDGES * 4);
    int*   dstB    = (int*)carve((size_t)N_EDGES * 4);
    float* ewB     = (float*)carve((size_t)N_EDGES * 4);

    // ---- edges for layer 0 (weight 1.0)
    init_edges<<<cdiv(N_EDGES, 256), 256, 0, stream>>>(ei, srcA, dstA, ewA, N_EDGES);

    // ---- layer 0: GCN(x, W0) + ReLU  -> nb1
    run_gcn(x, W0, b0, srcA, dstA, ewA, N0_NODES, 64, 1, nb0, agg, deg, nb1, stream);

    // ---- pool 0: N0 -> K1, edges A -> B, pooled x -> nb0
    run_pool(nb1, p0, N0_NODES, K1_NODES, srcA, dstA, ewA, srcB, dstB, ewB,
             score, mapping, bc, bo, st, nb0, stream);

    // ---- layer 1: GCN(nb0, W1) + ReLU -> nb0 (h staged in nb1)
    run_gcn(nb0, W1, b1, srcB, dstB, ewB, K1_NODES, 64, 1, nb1, agg, deg, nb0, stream);

    // ---- pool 1: K1 -> K2, edges B -> A, pooled x -> nb1
    run_pool(nb0, p1, K1_NODES, K2_NODES, srcB, dstB, ewB, srcA, dstA, ewA,
             score, mapping, bc, bo, st, nb1, stream);

    // ---- layer 2: GCN(nb1, W2) + ReLU -> nb1 (h staged in nb0)
    run_gcn(nb1, W2, b2, srcA, dstA, ewA, K2_NODES, 64, 1, nb0, agg, deg, nb1, stream);

    // ---- layer 3: GCN(nb1, W3) -> d_out [K2 x 32], no ReLU
    run_gcn(nb1, W3, b3, srcA, dstA, ewA, K2_NODES, 32, 0, nb0, agg, deg,
            (float*)d_out, stream);

    (void)in_sizes; (void)n_in; (void)out_size; (void)ws_size;
}
